// RGCNStack_2568390443228
// MI455X (gfx1250) — matrix operand order repaired and checked by reading, not compiled, not run on hardware
//
#include <hip/hip_runtime.h>
#include <hip/hip_bf16.h>

typedef __attribute__((ext_vector_type(16))) _Float16 v16h;
typedef __attribute__((ext_vector_type(8)))  _Float16 v8h;
typedef __attribute__((ext_vector_type(8)))  float    v8f;
typedef __attribute__((ext_vector_type(4)))  float    v4f;

#define N_NODES 100000
#define N_REL   16
#define N_BASES 12
#define DIM     128
#define N_EDGES 1600000

// ---------------------------------------------------------------------------
// Zero fill (16B granularity, grid-stride, nontemporal: S >> L2)
// ---------------------------------------------------------------------------
__global__ void zero_kernel(v4f* __restrict__ p, long long n4) {
    long long i = (long long)blockIdx.x * blockDim.x + threadIdx.x;
    long long stride = (long long)gridDim.x * blockDim.x;
    v4f z = {0.f, 0.f, 0.f, 0.f};
    for (; i < n4; i += stride) __builtin_nontemporal_store(z, &p[i]);
}

// ---------------------------------------------------------------------------
// Build W[r] = sum_b comp[r,b] * basis[b]  (and root as pseudo-relation 16),
// stored TRANSPOSED + f16:  wt[r][o][i]  so a WMMA B-fragment (K-major per
// output column) is two contiguous 16B loads per lane.
// ---------------------------------------------------------------------------
__global__ void build_w_kernel(const float* __restrict__ comp,   // [16,12]
                               const float* __restrict__ basis,  // [12,128,128]
                               const float* __restrict__ root,   // [128,128]
                               _Float16*   __restrict__ wt)      // [17,128,128] o-major
{
    int idx = blockIdx.x * blockDim.x + threadIdx.x;
    if (idx >= 17 * DIM * DIM) return;
    int r   = idx >> 14;
    int rem = idx & 16383;
    int i   = rem >> 7;   // input dim
    int o   = rem & 127;  // output dim
    float v;
    if (r < N_REL) {
        float acc = 0.f;
#pragma unroll
        for (int b = 0; b < N_BASES; ++b)
            acc += comp[r * N_BASES + b] * basis[(b << 14) + (i << 7) + o];
        v = acc;
    } else {
        v = root[(i << 7) + o];
    }
    wt[((size_t)r << 14) + (o << 7) + i] = (_Float16)v;
}

// ---------------------------------------------------------------------------
// Edge scatter: one wave per edge; each lane atomically adds 4 floats of the
// source row into S[tgt, etype, :].  Counts built only on the first layer.
// ---------------------------------------------------------------------------
__global__ void scatter_kernel(const float* __restrict__ x,      // [N,128]
                               const int*   __restrict__ src,
                               const int*   __restrict__ tgt,
                               const int*   __restrict__ etype,
                               float* __restrict__ S,            // [N,16,128]
                               float* __restrict__ cnt,          // [N,16]
                               int do_cnt)
{
    int lane = threadIdx.x & 31;
    int wave = threadIdx.x >> 5;
    long long e = (long long)blockIdx.x * 8 + wave;
    if (e >= N_EDGES) return;
    int s = src[e], t = tgt[e], r = etype[e];
    const v4f v = *reinterpret_cast<const v4f*>(x + (size_t)s * DIM + lane * 4);
    float* dst = S + (size_t)t * (N_REL * DIM) + (size_t)r * DIM + lane * 4;
    atomicAdd(dst + 0, v.x);
    atomicAdd(dst + 1, v.y);
    atomicAdd(dst + 2, v.z);
    atomicAdd(dst + 3, v.w);
    if (do_cnt && lane == 0) atomicAdd(cnt + (size_t)t * N_REL + r, 1.0f);
}

// ---------------------------------------------------------------------------
// Stage one A tile: mean[node0..node0+15, ir, :]  (or x rows for ir==16)
// converted to f16 in LDS.  Thread t handles node tid>>4, k-chunk (tid&15)*8:
// two NT b128 global loads + one v_rcp + one b128 LDS store.
// ---------------------------------------------------------------------------
__device__ __forceinline__ void stage_tile(
    _Float16 (*__restrict__ At)[DIM + 8], int ir, int node0, int tid,
    const float* __restrict__ S, const float* __restrict__ cnt,
    const float* __restrict__ x)
{
    const int nl   = tid >> 4;
    const int k0   = (tid & 15) << 3;
    const int node = node0 + nl;
    v4f a0, a1;
    float r;
    if (ir < N_REL) {
        float cv = cnt[(size_t)node * N_REL + ir];
        // Fast v_rcp_f32: result feeds an f16 convert, 1-ulp rcp is plenty.
        r = __builtin_amdgcn_rcpf(fmaxf(cv, 1.0f));
        const v4f* sp = reinterpret_cast<const v4f*>(
            S + (size_t)node * (N_REL * DIM) + (size_t)ir * DIM + k0);
        // S streams once (819MB >> L2): keep it out of the caches.
        a0 = __builtin_nontemporal_load(sp);
        a1 = __builtin_nontemporal_load(sp + 1);
    } else {
        r = 1.0f;
        const v4f* sp = reinterpret_cast<const v4f*>(x + (size_t)node * DIM + k0);
        a0 = sp[0];
        a1 = sp[1];
    }
    v8h h;
    h[0] = (_Float16)(a0.x * r); h[1] = (_Float16)(a0.y * r);
    h[2] = (_Float16)(a0.z * r); h[3] = (_Float16)(a0.w * r);
    h[4] = (_Float16)(a1.x * r); h[5] = (_Float16)(a1.y * r);
    h[6] = (_Float16)(a1.z * r); h[7] = (_Float16)(a1.w * r);
    *reinterpret_cast<v8h*>(&At[nl][k0]) = h;
}

// ---------------------------------------------------------------------------
// WMMA transform: out[n,:] = relu( sum_r mean[n,r,:] @ W[r] + x[n,:] @ root + bias )
// One block = 16 nodes x 128 outputs; 8 waves, each owns one 16x16 D tile.
// Double-buffered A tile in LDS (stage ir+1 while computing ir): 1 barrier/iter.
// K reduction: 17 relations x 128 = 2176, in 68 v_wmma_f32_16x16x32_f16 steps.
// ---------------------------------------------------------------------------
__global__ __launch_bounds__(256) void rgcn_gemm_kernel(
    const float*    __restrict__ S,     // [N,16,128]
    const float*    __restrict__ cnt,   // [N,16]
    const float*    __restrict__ x,     // [N,128] (root/self input)
    const _Float16* __restrict__ wt,    // [17,128,128] o-major f16
    const float*    __restrict__ bias,  // [128]
    float*          __restrict__ xout)  // [N,128]
{
    __shared__ __align__(16) _Float16 Atile[2][16][DIM + 8]; // +8 halfs: bank pad

    const int tid     = threadIdx.x;
    const int lane    = tid & 31;
    const int wave    = tid >> 5;         // 0..7 -> output column tile
    const int node0   = blockIdx.x * 16;
    const int colbase = wave * 16;
    const int row     = lane & 15;        // A-fragment row  (M)
    const int col     = lane & 15;        // B/D-fragment col (N)
    const int hi      = lane >> 4;        // lane half select
    const int kbaseA  = hi * 8;           // A: K offset within 32-chunk
    const int kbaseB  = hi * 8;           // B: K offset within 32-chunk

    v8f c = {};

    stage_tile(Atile[0], 0, node0, tid, S, cnt, x);
    __syncthreads();

    for (int ir = 0; ir < N_REL + 1; ++ir) {
        // Prefetch-stage the next relation's tile into the other buffer:
        // its global loads overlap the WMMA chain below.
        if (ir + 1 < N_REL + 1)
            stage_tile(Atile[(ir + 1) & 1], ir + 1, node0, tid, S, cnt, x);

        const _Float16(*__restrict__ At)[DIM + 8] = Atile[ir & 1];
        const _Float16* brow = wt + ((size_t)ir << 14) + (size_t)(colbase + col) * DIM;
#pragma unroll
        for (int ks = 0; ks < 4; ++ks) {
            const int kk = ks * 32;
            // A fragment: two contiguous 16B LDS reads (ISA 16-bit A layout)
            const v8h alo = *reinterpret_cast<const v8h*>(&At[row][kk + kbaseA]);
            const v8h ahi = *reinterpret_cast<const v8h*>(&At[row][kk + 16 + kbaseA]);
            v16h a = __builtin_shufflevector(alo, ahi,
                                             0, 1, 2, 3, 4, 5, 6, 7,
                                             8, 9, 10, 11, 12, 13, 14, 15);
            // B fragment: two contiguous 16B global reads (o-major weights, L2-hot)
            const v8h blo = *reinterpret_cast<const v8h*>(brow + kk + kbaseB);
            const v8h bhi = *reinterpret_cast<const v8h*>(brow + kk + 16 + kbaseB);
            v16h b = __builtin_shufflevector(blo, bhi,
                                             0, 1, 2, 3, 4, 5, 6, 7,
                                             8, 9, 10, 11, 12, 13, 14, 15);
            c = __builtin_amdgcn_wmma_f32_16x16x32_f16(
                    /*neg_a=*/false, a, /*neg_b=*/false, b,
                    /*c_mod=*/(short)0, c, /*reuse_a=*/false, /*reuse_b=*/false);
        }
        __syncthreads();
    }

    // Epilogue: D layout -> VGPR g holds row (g + 8*hi), col = lane&15.
    const float bv = bias[colbase + col];
#pragma unroll
    for (int g = 0; g < 8; ++g) {
        int r = g + hi * 8;
        float v = fmaxf(c[g] + bv, 0.0f);
        xout[(size_t)(node0 + r) * DIM + (colbase + col)] = v;
    }
}

// ---------------------------------------------------------------------------
// Final concat: out[n] = [x3 | x2 | x1 | emb]  (16B per thread, NT stores)
// ---------------------------------------------------------------------------
__global__ void concat_kernel(const float* __restrict__ x3, const float* __restrict__ x2,
                              const float* __restrict__ x1, const float* __restrict__ emb,
                              float* __restrict__ out)
{
    long long idx = (long long)blockIdx.x * blockDim.x + threadIdx.x;
    if (idx >= (long long)N_NODES * 128) return;   // v4f count: N * 512/4
    int n  = (int)(idx >> 7);
    int c4 = (int)(idx & 127);
    int seg = c4 >> 5;       // 0..3
    int w   = c4 & 31;       // v4f index within 128-float segment
    const float* srcs[4] = { x3, x2, x1, emb };
    const v4f v = reinterpret_cast<const v4f*>(srcs[seg] + (size_t)n * DIM)[w];
    __builtin_nontemporal_store(v, &reinterpret_cast<v4f*>(out + (size_t)n * 512)[c4]);
}

// ---------------------------------------------------------------------------
extern "C" void kernel_launch(void* const* d_in, const int* in_sizes, int n_in,
                              void* d_out, int out_size, void* d_ws, size_t ws_size,
                              hipStream_t stream) {
    const float* emb    = (const float*)d_in[0];
    const float* comp1  = (const float*)d_in[1];
    const float* basis1 = (const float*)d_in[2];
    const float* root1  = (const float*)d_in[3];
    const float* bias1  = (const float*)d_in[4];
    const float* comp2  = (const float*)d_in[5];
    const float* basis2 = (const float*)d_in[6];
    const float* root2  = (const float*)d_in[7];
    const float* bias2  = (const float*)d_in[8];
    const float* comp3  = (const float*)d_in[9];
    const float* basis3 = (const float*)d_in[10];
    const float* root3  = (const float*)d_in[11];
    const float* bias3  = (const float*)d_in[12];
    const int*   src    = (const int*)d_in[13];
    const int*   tgt    = (const int*)d_in[14];
    const int*   etype  = (const int*)d_in[15];
    float* out = (float*)d_out;

    // Workspace carve-up (all offsets 256B-aligned)
    char* ws = (char*)d_ws;
    const long long S_BYTES   = (long long)N_NODES * N_REL * DIM * 4;  // 819,200,000
    const long long CNT_BYTES = (long long)N_NODES * N_REL * 4;        //   6,400,000
    const long long X_BYTES   = (long long)N_NODES * DIM * 4;          //  51,200,000
    float*    S   = (float*)ws;
    float*    cnt = (float*)(ws + S_BYTES);
    float*    x1  = (float*)(ws + S_BYTES + CNT_BYTES);
    float*    x2  = (float*)(ws + S_BYTES + CNT_BYTES + X_BYTES);
    float*    x3  = (float*)(ws + S_BYTES + CNT_BYTES + 2 * X_BYTES);
    _Float16* wt  = (_Float16*)(ws + S_BYTES + CNT_BYTES + 3 * X_BYTES);

    const int ZB = 8192, ZT = 256;
    const int WBLK = (17 * DIM * DIM + 255) / 256;
    const int SBLK = N_EDGES / 8;          // 200,000 (one wave per edge)
    const int GBLK = N_NODES / 16;         // 6,250
    const int CBLK = (int)(((long long)N_NODES * 128 + 255) / 256);

    // ---- Layer 1 (zero S + cnt together: contiguous) ----
    zero_kernel<<<ZB, ZT, 0, stream>>>((v4f*)S, (S_BYTES + CNT_BYTES) / 16);
    build_w_kernel<<<WBLK, 256, 0, stream>>>(comp1, basis1, root1, wt);
    scatter_kernel<<<SBLK, 256, 0, stream>>>(emb, src, tgt, etype, S, cnt, 1);
    rgcn_gemm_kernel<<<GBLK, 256, 0, stream>>>(S, cnt, emb, wt, bias1, x1);

    // ---- Layer 2 ----
    zero_kernel<<<ZB, ZT, 0, stream>>>((v4f*)S, S_BYTES / 16);
    build_w_kernel<<<WBLK, 256, 0, stream>>>(comp2, basis2, root2, wt);
    scatter_kernel<<<SBLK, 256, 0, stream>>>(x1, src, tgt, etype, S, cnt, 0);
    rgcn_gemm_kernel<<<GBLK, 256, 0, stream>>>(S, cnt, x1, wt, bias2, x2);

    // ---- Layer 3 ----
    zero_kernel<<<ZB, ZT, 0, stream>>>((v4f*)S, S_BYTES / 16);
    build_w_kernel<<<WBLK, 256, 0, stream>>>(comp3, basis3, root3, wt);
    scatter_kernel<<<SBLK, 256, 0, stream>>>(x2, src, tgt, etype, S, cnt, 0);
    rgcn_gemm_kernel<<<GBLK, 256, 0, stream>>>(S, cnt, x2, wt, bias3, x3);

    // ---- Concat [x3|x2|x1|emb] ----
    concat_kernel<<<CBLK, 256, 0, stream>>>(x3, x2, x1, emb, out);
}
